// SparseAttentionWithVisualization_69853348102277
// MI455X (gfx1250) — matrix operand order repaired and checked by reading, not compiled, not run on hardware
//
#include <hip/hip_runtime.h>

#define Bc 4
#define Sc 2048
#define Dc 1024
#define BSc 8192   // B*S

typedef __attribute__((ext_vector_type(16))) __bf16          v16bf;
typedef __attribute__((ext_vector_type(16))) unsigned short  v16u;
typedef __attribute__((ext_vector_type(8)))  unsigned short  v8u;
typedef __attribute__((ext_vector_type(8)))  float           v8f;
typedef __attribute__((ext_vector_type(4)))  float           f32x4;
typedef __attribute__((ext_vector_type(2)))  unsigned int    u32x2;
typedef __attribute__((ext_vector_type(4)))  unsigned int    u32x4;
typedef __attribute__((ext_vector_type(4)))  int             i32x4;

__device__ __forceinline__ unsigned short f32_bf16(float f) {
  unsigned u = __float_as_uint(f);
  unsigned r = u + 0x7FFFu + ((u >> 16) & 1u);   // round-to-nearest-even
  return (unsigned short)(r >> 16);
}
__device__ __forceinline__ unsigned pack2(float a, float b) {
  return (unsigned)f32_bf16(a) | ((unsigned)f32_bf16(b) << 16);
}
// Build a 16-element bf16 fragment from two contiguous 8x16-bit LDS chunks.
__device__ __forceinline__ v16bf frag_from(const unsigned short* p0,
                                           const unsigned short* p1) {
  v8u lo = *(const v8u*)p0;
  v8u hi = *(const v8u*)p1;
  v16u u = __builtin_shufflevector(lo, hi, 0,1,2,3,4,5,6,7,8,9,10,11,12,13,14,15);
  return __builtin_bit_cast(v16bf, u);
}

// 16-byte global->LDS copy: async path (ASYNCcnt) when toolchain exposes it.
#if defined(__has_builtin)
#if __has_builtin(__builtin_amdgcn_global_load_async_to_lds_b128)
#define HAVE_ASYNC_LDS 1
#endif
#endif

__device__ __forceinline__ void copy16_to_lds(unsigned short* ldst,
                                              const unsigned short* gsrc) {
#ifdef HAVE_ASYNC_LDS
  __builtin_amdgcn_global_load_async_to_lds_b128(
      (__attribute__((address_space(1))) i32x4*)(gsrc),
      (__attribute__((address_space(3))) i32x4*)(ldst), 0, 0);
#else
  *(u32x4*)ldst = *(const u32x4*)gsrc;
#endif
}
template <int N>
__device__ __forceinline__ void async_wait() {
#ifdef HAVE_ASYNC_LDS
#if __has_builtin(__builtin_amdgcn_s_wait_asynccnt)
  __builtin_amdgcn_s_wait_asynccnt(N);
#else
  asm volatile("s_wait_asynccnt %0" :: "i"(N) : "memory");
#endif
#endif
}

// ---------------------------------------------------------------------------
// f32 -> bf16 row-major conversion (n must be multiple of 1024).
// ---------------------------------------------------------------------------
__global__ __launch_bounds__(256)
void convert_bf16(const float* __restrict__ src, unsigned short* __restrict__ dst) {
  size_t c = (size_t)blockIdx.x * 256 + threadIdx.x;
  f32x4 v = *(const f32x4*)&src[c * 4];
  u32x2 p; p.x = pack2(v.x, v.y); p.y = pack2(v.z, v.w);
  *(u32x2*)&dst[c * 4] = p;
}

// ---------------------------------------------------------------------------
// Weight transpose + convert: W (K x N, f32) -> Wt (N x K, bf16). 64x64 tiles.
// ---------------------------------------------------------------------------
__global__ __launch_bounds__(256)
void transpose_convert_bf16(const float* __restrict__ W,
                            unsigned short* __restrict__ Wt, int K, int N) {
  __shared__ float ld[64][65];
  const int tid = threadIdx.x;
  const int tile_k = blockIdx.y * 64;
  const int tile_n = blockIdx.x * 64;
  #pragma unroll
  for (int i = 0; i < 4; ++i) {
    int c = tid + i * 256;
    int k = c >> 4, nb = (c & 15) * 4;
    f32x4 v = *(const f32x4*)&W[(size_t)(tile_k + k) * N + tile_n + nb];
    ld[k][nb] = v.x; ld[k][nb + 1] = v.y; ld[k][nb + 2] = v.z; ld[k][nb + 3] = v.w;
  }
  __syncthreads();
  #pragma unroll
  for (int i = 0; i < 2; ++i) {
    int c = tid + i * 256;
    int n = c >> 3, kb = (c & 7) * 8;
    unsigned dw[4];
    #pragma unroll
    for (int u = 0; u < 4; ++u)
      dw[u] = pack2(ld[kb + 2 * u][n], ld[kb + 2 * u + 1][n]);
    u32x4 p = {dw[0], dw[1], dw[2], dw[3]};
    *(u32x4*)&Wt[(size_t)(tile_n + n) * K + tile_k + kb] = p;
  }
}

// ---------------------------------------------------------------------------
// GEMM: C(M,N) = A(M,K) * Bt(N,K)^T + bias; A,Bt bf16 row-major, C f32.
// 64x64x64 tiles; double-buffered async-to-LDS staging pipelined with WMMA.
// ---------------------------------------------------------------------------
__global__ __launch_bounds__(256)
void wmma_gemm_bias(const unsigned short* __restrict__ A,
                    const unsigned short* __restrict__ Bt,
                    const float* __restrict__ bias, float* __restrict__ C,
                    int M, int N, int K) {
  __shared__ alignas(16) unsigned short lA [2][64][64];   // [buf][m][k]
  __shared__ alignas(16) unsigned short lBt[2][64][64];   // [buf][n][k]

  const int tid  = threadIdx.x;
  const int wave = tid >> 5;
  const int lane = tid & 31;
  const int wm   = wave & 3;
  const int wn   = wave >> 2;
  const int m16  = lane & 15;
  const int hi   = lane >> 4;

  const int tile_m = blockIdx.y * 64;
  const int tile_n = blockIdx.x * 64;

  const int ar = wm * 16 + m16;
  const int n0 = wn * 32 + m16;
  const int n1 = n0 + 16;

  const int cr = tid >> 3;             // staged row (0..31) per chunk-set
  const int ck = (tid & 7) * 8;        // staged k offset

  auto stage = [&](int k0, int p) {    // 4 async copies per thread/wave
    #pragma unroll
    for (int i = 0; i < 2; ++i) {
      int r = cr + i * 32;
      copy16_to_lds(&lA [p][r][ck], &A [(size_t)(tile_m + r) * K + k0 + ck]);
      copy16_to_lds(&lBt[p][r][ck], &Bt[(size_t)(tile_n + r) * K + k0 + ck]);
    }
  };

  v8f acc0 = {}; v8f acc1 = {};

  const int T = K / 64;
  stage(0, 0);
  for (int it = 0; it < T; ++it) {
    const int p = it & 1;
    const int k0 = it * 64;
    if (it + 1 < T) {
      stage(k0 + 64, 1 - p);           // prefill next buffer (async)
      if (it + 2 < T) {                // warm L2 two tiles ahead
        __builtin_prefetch(&A [(size_t)(tile_m + (tid & 63)) * K + k0 + 128], 0, 3);
        __builtin_prefetch(&Bt[(size_t)(tile_n + (tid & 63)) * K + k0 + 128], 0, 3);
      }
      async_wait<4>();                 // current tile's 4 copies complete
    } else {
      async_wait<0>();
    }
    __syncthreads();

    #pragma unroll
    for (int ks = 0; ks < 64; ks += 32) {
      v16bf a  = frag_from(&lA[p][ar][ks + hi * 16], &lA[p][ar][ks + hi * 16 + 8]);
      v16bf b0 = frag_from(&lBt[p][n0][ks + hi * 16], &lBt[p][n0][ks + hi * 16 + 8]);
      v16bf b1 = frag_from(&lBt[p][n1][ks + hi * 16], &lBt[p][n1][ks + hi * 16 + 8]);
      acc0 = __builtin_amdgcn_wmma_f32_16x16x32_bf16(false, a, false, b0, (short)0, acc0, false, false);
      acc1 = __builtin_amdgcn_wmma_f32_16x16x32_bf16(false, a, false, b1, (short)0, acc1, false, false);
    }
    __syncthreads();                   // buf p restaged by next iteration
  }

  #pragma unroll
  for (int i = 0; i < 8; ++i) {
    int row  = tile_m + wm * 16 + i + hi * 8;
    int col0 = tile_n + wn * 32 + m16;
    int col1 = col0 + 16;
    C[(size_t)row * N + col0] = acc0[i] + bias[col0];
    C[(size_t)row * N + col1] = acc1[i] + bias[col1];
  }
}

// ---------------------------------------------------------------------------
// Tiny projection iw = x @ w_iw + b_iw (N=4): one wave per (b,t) row.
// ---------------------------------------------------------------------------
__global__ __launch_bounds__(256)
void proj_iw(const float* __restrict__ x, const float* __restrict__ w,
             const float* __restrict__ bias, float* __restrict__ out) {
  int wave = threadIdx.x >> 5, lane = threadIdx.x & 31;
  size_t bt = (size_t)blockIdx.x * 8 + wave;
  const float* xr = x + bt * Dc;
  float a0 = 0.f, a1 = 0.f, a2 = 0.f, a3 = 0.f;
  #pragma unroll 4
  for (int u = 0; u < 32; ++u) {
    int d = u * 32 + lane;
    float xv = xr[d];
    const float* wr = w + (size_t)d * 4;
    a0 += xv * wr[0]; a1 += xv * wr[1]; a2 += xv * wr[2]; a3 += xv * wr[3];
  }
  #pragma unroll
  for (int o = 16; o > 0; o >>= 1) {
    a0 += __shfl_xor(a0, o, 32); a1 += __shfl_xor(a1, o, 32);
    a2 += __shfl_xor(a2, o, 32); a3 += __shfl_xor(a3, o, 32);
  }
  if (lane == 0) {
    float* o4 = out + bt * 4;
    o4[0] = a0 + bias[0]; o4[1] = a1 + bias[1];
    o4[2] = a2 + bias[2]; o4[3] = a3 + bias[3];
  }
}

// ---------------------------------------------------------------------------
// RoPE in-place on q and k slices of qkv (B,S,3072); half = 512.
// ---------------------------------------------------------------------------
__global__ __launch_bounds__(256)
void rope_qk(float* __restrict__ qkv) {
  size_t gid = (size_t)blockIdx.x * 256 + threadIdx.x;  // < BS*512
  int    i   = (int)(gid & 511);
  size_t bt  = gid >> 9;
  int    t   = (int)(bt & (Sc - 1));
  float inv_freq = powf(10000.f, -(float)i / 512.f);
  float ang = (float)t * inv_freq;
  float cs = cosf(ang), sn = sinf(ang);
  float* base = qkv + bt * 3072;
  float x1 = base[i], x2 = base[i + 512];
  base[i]       = x1 * cs - x2 * sn;
  base[i + 512] = x1 * sn + x2 * cs;
  float y1 = base[1024 + i], y2 = base[1024 + i + 512];
  base[1024 + i]       = y1 * cs - y2 * sn;
  base[1024 + i + 512] = y1 * sn + y2 * cs;
}

// ---------------------------------------------------------------------------
// Indexer: combined[b,t,s] = sum_h relu(iq[b,t,h,:].ik[b,s,:]) * iw[b,t,h]
// bf16 inputs; ik tile staged once, iq tile double-buffered across heads.
// ---------------------------------------------------------------------------
__global__ __launch_bounds__(256)
void indexer_scores(const unsigned short* __restrict__ iqb,
                    const unsigned short* __restrict__ ikb,
                    const float* __restrict__ iw, float* __restrict__ comb) {
  __shared__ alignas(16) unsigned short lA [2][64][64];  // iq tile (head h)
  __shared__ alignas(16) unsigned short lBt[64][64];     // ik tile [s][k]
  __shared__ float lW[64][4];

  const int tid  = threadIdx.x;
  const int wave = tid >> 5;
  const int lane = tid & 31;
  const int wm   = wave & 3;
  const int wn   = wave >> 2;
  const int m16  = lane & 15;
  const int hi   = lane >> 4;

  const int b      = blockIdx.z;
  const int tile_t = blockIdx.y * 64;
  const int tile_s = blockIdx.x * 64;

  const int ar = wm * 16 + m16;
  const int n0 = wn * 32 + m16;
  const int n1 = n0 + 16;

  const int cr = tid >> 3;
  const int ck = (tid & 7) * 8;

  auto stage_iq = [&](int h, int p) {  // 2 async copies per thread/wave
    #pragma unroll
    for (int i = 0; i < 2; ++i) {
      int r = cr + i * 32;
      copy16_to_lds(&lA[p][r][ck],
                    &iqb[((size_t)b * Sc + tile_t + r) * 256 + h * 64 + ck]);
    }
  };

  // stage ik tile once (2 async copies), then head 0's iq
  #pragma unroll
  for (int i = 0; i < 2; ++i) {
    int r = cr + i * 32;
    copy16_to_lds(&lBt[r][ck], &ikb[((size_t)b * Sc + tile_s + r) * 64 + ck]);
  }
  stage_iq(0, 0);
  { int r = tid >> 2, h = tid & 3;
    lW[r][h] = iw[((size_t)b * Sc + tile_t + r) * 4 + h]; }

  v8f acc0 = {}; v8f acc1 = {};

  for (int h = 0; h < 4; ++h) {
    const int p = h & 1;
    if (h + 1 < 4) { stage_iq(h + 1, 1 - p); async_wait<2>(); }
    else           { async_wait<0>(); }
    __syncthreads();

    v8f h0 = {}; v8f h1 = {};
    #pragma unroll
    for (int ks = 0; ks < 64; ks += 32) {
      v16bf a  = frag_from(&lA[p][ar][ks + hi * 16], &lA[p][ar][ks + hi * 16 + 8]);
      v16bf b0 = frag_from(&lBt[n0][ks + hi * 16], &lBt[n0][ks + hi * 16 + 8]);
      v16bf b1 = frag_from(&lBt[n1][ks + hi * 16], &lBt[n1][ks + hi * 16 + 8]);
      h0 = __builtin_amdgcn_wmma_f32_16x16x32_bf16(false, a, false, b0, (short)0, h0, false, false);
      h1 = __builtin_amdgcn_wmma_f32_16x16x32_bf16(false, a, false, b1, (short)0, h1, false, false);
    }
    #pragma unroll
    for (int i = 0; i < 8; ++i) {
      int rowl = wm * 16 + i + hi * 8;
      float w = lW[rowl][h];
      acc0[i] += fmaxf(h0[i], 0.f) * w;
      acc1[i] += fmaxf(h1[i], 0.f) * w;
    }
    __syncthreads();
  }

  #pragma unroll
  for (int i = 0; i < 8; ++i) {
    int rowl = wm * 16 + i + hi * 8;
    float* crow = comb + ((size_t)b * Sc + tile_t + rowl) * Sc;
    crow[tile_s + wn * 32 + m16]      = acc0[i];
    crow[tile_s + wn * 32 + 16 + m16] = acc1[i];
  }
}

// ---------------------------------------------------------------------------
// Top-k: per (b,t) row of combined -> indices of top 256 (bitonic, LDS).
// ---------------------------------------------------------------------------
__global__ __launch_bounds__(256)
void topk_select(const float* __restrict__ comb, int* __restrict__ sel) {
  __shared__ float          val[Sc];
  __shared__ unsigned short idx[Sc];
  const int b = blockIdx.y, t = blockIdx.x;
  const float* row = comb + ((size_t)b * Sc + t) * Sc;
  for (int i = threadIdx.x; i < Sc; i += 256) { val[i] = row[i]; idx[i] = (unsigned short)i; }
  __syncthreads();
  for (int k = 2; k <= Sc; k <<= 1) {
    for (int j = k >> 1; j > 0; j >>= 1) {
      for (int n = threadIdx.x; n < Sc; n += 256) {
        int p = n ^ j;
        if (p > n) {
          bool desc = ((n & k) == 0);
          float a = val[n], c = val[p];
          if (desc ? (a < c) : (a > c)) {
            val[n] = c; val[p] = a;
            unsigned short tm = idx[n]; idx[n] = idx[p]; idx[p] = tm;
          }
        }
      }
      __syncthreads();
    }
  }
  sel[((size_t)b * Sc + t) * 256 + threadIdx.x] = (int)idx[threadIdx.x];
}

// ---------------------------------------------------------------------------
// Sparse attention over the 256 selected keys; emits bf16 for final GEMM.
// ---------------------------------------------------------------------------
__global__ __launch_bounds__(256)
void sparse_attn(const float* __restrict__ qkv, const int* __restrict__ sel,
                 unsigned short* __restrict__ outb) {
  __shared__ float qs[Dc];
  __shared__ float sc[256];
  __shared__ int   ss[256];
  __shared__ float red[256];
  const int b = blockIdx.y, t = blockIdx.x;
  const int tid = threadIdx.x;
  const int wave = tid >> 5, lane = tid & 31;

  const float* qrow = qkv + ((size_t)b * Sc + t) * 3072;
  *(f32x4*)&qs[4 * tid] = *(const f32x4*)&qrow[4 * tid];
  ss[tid] = sel[((size_t)b * Sc + t) * 256 + tid];
  __syncthreads();

  for (int jj = 0; jj < 32; ++jj) {
    int j = wave * 32 + jj;
    int s = ss[j];
    const float* krow = qkv + ((size_t)b * Sc + s) * 3072 + 1024;
    float p = 0.f;
    #pragma unroll 8
    for (int u = 0; u < 32; ++u) {
      int d = u * 32 + lane;
      p += qs[d] * krow[d];
    }
    #pragma unroll
    for (int o = 16; o > 0; o >>= 1) p += __shfl_xor(p, o, 32);
    if (lane == 0) sc[j] = p * 0.03125f;   // 1/sqrt(1024)
  }
  __syncthreads();

  float myv = sc[tid];
  red[tid] = myv; __syncthreads();
  #pragma unroll
  for (int o = 128; o > 0; o >>= 1) { if (tid < o) red[tid] = fmaxf(red[tid], red[tid + o]); __syncthreads(); }
  float mx = red[0]; __syncthreads();
  float e = expf(myv - mx);
  red[tid] = e; __syncthreads();
  #pragma unroll
  for (int o = 128; o > 0; o >>= 1) { if (tid < o) red[tid] += red[tid + o]; __syncthreads(); }
  float inv = 1.f / red[0];
  __syncthreads();
  sc[tid] = e * inv;
  __syncthreads();

  float acc[4] = {0.f, 0.f, 0.f, 0.f};
  for (int j = 0; j < 256; ++j) {
    int s = ss[j]; float p = sc[j];
    const float* vrow = qkv + ((size_t)b * Sc + s) * 3072 + 2048;
    f32x4 v = *(const f32x4*)&vrow[4 * tid];
    acc[0] += p * v.x; acc[1] += p * v.y; acc[2] += p * v.z; acc[3] += p * v.w;
  }
  u32x2 pk; pk.x = pack2(acc[0], acc[1]); pk.y = pack2(acc[2], acc[3]);
  *(u32x2*)&outb[((size_t)b * Sc + t) * Dc + 4 * tid] = pk;
}

// ---------------------------------------------------------------------------
extern "C" void kernel_launch(void* const* d_in, const int* in_sizes, int n_in,
                              void* d_out, int out_size, void* d_ws, size_t ws_size,
                              hipStream_t stream) {
  const float* x     = (const float*)d_in[0];
  const float* w_qkv = (const float*)d_in[1];
  const float* b_qkv = (const float*)d_in[2];
  const float* w_o   = (const float*)d_in[3];
  const float* b_o   = (const float*)d_in[4];
  const float* w_iq  = (const float*)d_in[5];
  const float* b_iq  = (const float*)d_in[6];
  const float* w_ik  = (const float*)d_in[7];
  const float* b_ik  = (const float*)d_in[8];
  const float* w_iw  = (const float*)d_in[9];
  const float* b_iw  = (const float*)d_in[10];

  // ---- bf16 pool ----
  unsigned short* xb    = (unsigned short*)d_ws;            // BS x 1024
  unsigned short* wqkvT = xb    + (size_t)BSc * Dc;         // 3072 x 1024
  unsigned short* wiqT  = wqkvT + (size_t)3072 * Dc;        // 256 x 1024
  unsigned short* wikT  = wiqT  + (size_t)256 * Dc;         // 64 x 1024
  unsigned short* woT   = wikT  + (size_t)64 * Dc;          // 1024 x 1024
  unsigned short* iqb16 = woT   + (size_t)Dc * Dc;          // BS x 256
  unsigned short* ikb16 = iqb16 + (size_t)BSc * 256;        // BS x 64
  unsigned short* aoutb = ikb16 + (size_t)BSc * 64;         // BS x 1024
  // ---- f32 / int pool ----
  float* qkv  = (float*)(aoutb + (size_t)BSc * Dc);         // BS x 3072
  float* iq   = qkv  + (size_t)BSc * 3072;                  // BS x 256
  float* ikf  = iq   + (size_t)BSc * 256;                   // BS x 64
  float* iwb  = ikf  + (size_t)BSc * 64;                    // BS x 4
  float* comb = iwb  + (size_t)BSc * 4;                     // B x S x S
  int*   sel  = (int*)(comb + (size_t)Bc * Sc * Sc);        // BS x 256

  dim3 blk(256);
  // one-time bf16 conversions
  convert_bf16<<<dim3((BSc * Dc) / 1024), blk, 0, stream>>>(x, xb);
  transpose_convert_bf16<<<dim3(3072 / 64, Dc / 64), blk, 0, stream>>>(w_qkv, wqkvT, Dc, 3072);
  transpose_convert_bf16<<<dim3(256  / 64, Dc / 64), blk, 0, stream>>>(w_iq,  wiqT,  Dc, 256);
  transpose_convert_bf16<<<dim3(64   / 64, Dc / 64), blk, 0, stream>>>(w_ik,  wikT,  Dc, 64);
  transpose_convert_bf16<<<dim3(Dc   / 64, Dc / 64), blk, 0, stream>>>(w_o,   woT,   Dc, Dc);
  // projections
  wmma_gemm_bias<<<dim3(3072 / 64, BSc / 64), blk, 0, stream>>>(xb, wqkvT, b_qkv, qkv, BSc, 3072, Dc);
  wmma_gemm_bias<<<dim3(256  / 64, BSc / 64), blk, 0, stream>>>(xb, wiqT,  b_iq,  iq,  BSc, 256,  Dc);
  wmma_gemm_bias<<<dim3(64   / 64, BSc / 64), blk, 0, stream>>>(xb, wikT,  b_ik,  ikf, BSc, 64,   Dc);
  proj_iw<<<dim3(BSc / 8), blk, 0, stream>>>(x, w_iw, b_iw, iwb);
  rope_qk<<<dim3((BSc * 512) / 256), blk, 0, stream>>>(qkv);
  // indexer inputs to bf16, indexer, top-k
  convert_bf16<<<dim3((BSc * 256) / 1024), blk, 0, stream>>>(iq,  iqb16);
  convert_bf16<<<dim3((BSc * 64)  / 1024), blk, 0, stream>>>(ikf, ikb16);
  indexer_scores<<<dim3(Sc / 64, Sc / 64, Bc), blk, 0, stream>>>(iqb16, ikb16, iwb, comb);
  topk_select<<<dim3(Sc, Bc), blk, 0, stream>>>(comb, sel);
  // sparse attention + output projection
  sparse_attn<<<dim3(Sc, Bc), blk, 0, stream>>>(qkv, sel, aoutb);
  wmma_gemm_bias<<<dim3(Dc / 64, BSc / 64), blk, 0, stream>>>(aoutb, woT, b_o, (float*)d_out, BSc, Dc, Dc);
}
